// GptOssExperts_1176821039989
// MI455X (gfx1250) — compile-verified, matrix operand order checked by reading, not run on hardware
//
#include <hip/hip_runtime.h>
#include <hip/hip_bf16.h>

// ---------------- problem constants (from setup_inputs) ----------------
constexpr int T_   = 1024;
constexpr int H_   = 2880;
constexpr int I_   = 2880;
constexpr int E_   = 8;
constexpr int KSEL = 2;
constexpr int N1   = 2 * I_;       // 5760 gate_up rows
constexpr int NA   = T_ * KSEL;    // 2048 assignments (slots)
constexpr int NKB  = H_ / 32;      // 90 k-blocks (same for both GEMMs)

// ---------------- workspace layout (bytes) ----------------
constexpr size_t OFF_COUNTS = 0;
constexpr size_t OFF_OFFS   = 64;
constexpr size_t OFF_TOK    = 1024;                               // tokOfSlot[NA]
constexpr size_t OFF_SLOT   = OFF_TOK + (size_t)NA * 4;           // slotOf[NA]
constexpr size_t OFF_XQ     = 32768;                              // bf16 [T][H]
constexpr size_t OFF_H      = OFF_XQ  + (size_t)T_ * H_ * 2;      // f32 [NA][N1]
constexpr size_t OFF_ACT    = OFF_H   + (size_t)NA * N1 * 4;      // bf16 [NA][I]
constexpr size_t OFF_OA     = OFF_ACT + (size_t)NA * I_ * 2;      // f32 [NA][H]

// ---------------- WMMA types ----------------
typedef __bf16 bf16_t;
typedef __attribute__((ext_vector_type(16))) __bf16 v16bf;
typedef __attribute__((ext_vector_type(8)))  float  v8f;

union BF16x16 {
    v16bf v;
    unsigned u[8];
    uint4 q[2];
};

// ---------------- helpers ----------------
__device__ __forceinline__ unsigned short f32_to_bf16_bits(float f) {
    // values here are exactly representable in bf16 (fp8/fp4 mantissas * 2^e)
    return (unsigned short)(__float_as_uint(f) >> 16);
}

// quantize-dequantize |a|<=1 to fp8 e4m3 (RNE), returns float
__device__ __forceinline__ float qdq_e4m3(float a) {
    float aa = fabsf(a);
    float r;
    if (aa < 0.015625f) {                 // < 2^-6 : subnormal grid, step 2^-9
        r = rintf(aa * 512.0f) * (1.0f / 512.0f);
    } else {
        int ex; float fr = frexpf(aa, &ex);   // aa = fr*2^ex, fr in [0.5,1)
        r = rintf(ldexpf(fr, 4)) * ldexpf(1.0f, ex - 4);  // 4-bit significand RNE
    }
    return copysignf(r, a);
}

// ceil(log2(max(m,1e-12))) clipped to [-127,128], return 2^e
__device__ __forceinline__ float mx_scale(float m) {
    m = fmaxf(m, 1e-12f);
    int ex; float fr = frexpf(m, &ex);
    int ce = (fr == 0.5f) ? (ex - 1) : ex;
    ce = max(-127, min(128, ce));
    return ldexpf(1.0f, ce);
}

// Dequant 4 packed fp4-pair bytes (in one uint, byte k = pair k) into 4 uints,
// each uint = two bf16 (lo nibble -> low half, hi nibble -> high half).
// Pure bit logic via v_perm_b32 byte LUTs; NO scaling (scale applied later).
__device__ __forceinline__ void dq4(unsigned v4, unsigned* o) {
    unsigned bl = v4 & 0x07070707u;          // low-nibble magnitude codes
    unsigned bh = (v4 >> 4) & 0x07070707u;   // high-nibble magnitude codes
    // bf16 high byte table for mags 0..7: {00,3F,3F,3F,40,40,40,40}
    unsigned hL = __builtin_amdgcn_perm(0x40404040u, 0x3F3F3F00u, bl) |
                  ((v4 & 0x08080808u) << 4);                 // + sign of lo nibble
    unsigned hH = __builtin_amdgcn_perm(0x40404040u, 0x3F3F3F00u, bh) |
                  (v4 & 0x80808080u);                        // + sign of hi nibble
    // bf16 low byte table for mags 0..7: {00,00,80,C0,00,40,80,C0}
    unsigned lL = __builtin_amdgcn_perm(0xC0804000u, 0xC0800000u, bl);
    unsigned lH = __builtin_amdgcn_perm(0xC0804000u, 0xC0800000u, bh);
    // interleave bytes: out_k = {lL.bk, hL.bk, lH.bk, hH.bk}
    unsigned p01L = __builtin_amdgcn_perm(hL, lL, 0x05010400u); // {lL0,hL0,lL1,hL1}
    unsigned p01H = __builtin_amdgcn_perm(hH, lH, 0x05010400u);
    unsigned p23L = __builtin_amdgcn_perm(hL, lL, 0x07030602u); // {lL2,hL2,lL3,hL3}
    unsigned p23H = __builtin_amdgcn_perm(hH, lH, 0x07030602u);
    o[0] = (p01L & 0xFFFFu) | (p01H << 16);
    o[1] = (p01L >> 16)     | (p01H & 0xFFFF0000u);
    o[2] = (p23L & 0xFFFFu) | (p23H << 16);
    o[3] = (p23L >> 16)     | (p23H & 0xFFFF0000u);
}

// ---------------- kernel 1: routing (single block) ----------------
__global__ void route_kernel(const int* __restrict__ eidx,
                             int* __restrict__ counts, int* __restrict__ offs,
                             int* __restrict__ tokOfSlot, int* __restrict__ slotOf) {
    __shared__ int scnt[E_], soff[E_], scur[E_];
    int t = threadIdx.x;
    if (t < E_) { scnt[t] = 0; scur[t] = 0; }
    __syncthreads();
    for (int i = t; i < NA; i += blockDim.x) atomicAdd(&scnt[eidx[i]], 1);
    __syncthreads();
    if (t == 0) { int a = 0; for (int e = 0; e < E_; e++) { soff[e] = a; a += scnt[e]; } }
    __syncthreads();
    for (int i = t; i < NA; i += blockDim.x) {
        int e = eidx[i];
        int p = atomicAdd(&scur[e], 1);
        int slot = soff[e] + p;
        tokOfSlot[slot] = i >> 1;       // token index (i = t*2+k)
        slotOf[i] = slot;
    }
    if (t < E_) { counts[t] = scnt[t]; offs[t] = soff[t]; }
}

// ---------------- kernel 2: MXFP8 qdq of hidden_states -> bf16 ----------------
__global__ void qdqx_kernel(const float* __restrict__ x, unsigned short* __restrict__ xq) {
    int lane = threadIdx.x & 31;
    long blk = (long)blockIdx.x * (blockDim.x >> 5) + (threadIdx.x >> 5);
    if (blk >= (long)T_ * NKB) return;
    long idx = blk * 32 + lane;
    float v = x[idx];
    float m = fabsf(v);
    for (int d = 16; d > 0; d >>= 1) m = fmaxf(m, __shfl_xor(m, d, 32));
    float scale = mx_scale(m);
    float q = qdq_e4m3(v / scale) * scale;
    xq[idx] = f32_to_bf16_bits(q);
}

// ---------------- kernel 3/5: grouped WMMA GEMM ----------------
// Block: 256 threads = 8 waves. Tile: M=128 x N=128 (wave = 16 N cols, 8 M subtiles).
// A: bf16 rows (xq via token index, or act via slot). B: fp4 blocks + E8M0 scales.
// acc += 2^(scale-127) * wmma_bf16(A_tile, B_unscaled, 0)    (scale uniform per lane)
__global__ void __launch_bounds__(256)
moe_gemm(const unsigned short* __restrict__ Abase,
         const int* __restrict__ tokOfSlot, const int* __restrict__ offs,
         const int* __restrict__ cnts,
         const int* __restrict__ blocks, const int* __restrict__ scales,
         const float* __restrict__ bias, float* __restrict__ out,
         int N, int Kd, int useTok) {
    __shared__ __align__(16) unsigned ldsA[128 * 16];   // 128 rows x 32 bf16
    const int e = blockIdx.z;
    const int nE = cnts[e];
    const int mtile = blockIdx.y;
    if (mtile * 128 >= nE) return;
    const int offe = offs[e];
    const int lane = threadIdx.x & 31;
    const int wave = threadIdx.x >> 5;
    const int h    = lane >> 4;       // half (0/1)
    const int ln   = lane & 15;
    const int nkb  = Kd >> 5;

    // A staging assignment: thread -> (row rr, uint group c8)
    const int rr = threadIdx.x >> 1;
    const int c8 = (threadIdx.x & 1) * 8;
    int srow = mtile * 128 + rr; if (srow >= nE) srow = nE - 1;
    int slotr = offe + srow;
    long arow = useTok ? (long)tokOfSlot[slotr] : (long)slotr;
    const unsigned* Agr = (const unsigned*)Abase + arow * (long)(Kd >> 1);

    // B row for this lane (output column j)
    int j = blockIdx.x * 128 + wave * 16 + ln;
    int jc = min(j, N - 1);
    const int* Brow = blocks + ((long)e * N + jc) * (long)nkb * 16;
    const int* Srow = scales + ((long)e * N + jc) * (long)nkb;

    v8f acc[8];
    #pragma unroll
    for (int mt = 0; mt < 8; ++mt)
        acc[mt] = (v8f){0.f,0.f,0.f,0.f,0.f,0.f,0.f,0.f};
    const v8f zf = (v8f){0.f,0.f,0.f,0.f,0.f,0.f,0.f,0.f};

    for (int kb = 0; kb < nkb; ++kb) {
        __syncthreads();
        // stage A tile: 128 rows x 16 uints
        {
            const uint4* g = (const uint4*)(Agr + (long)kb * 16 + c8);
            uint4 a0 = g[0], a1 = g[1];
            *(uint4*)&ldsA[rr * 16 + c8]     = a0;
            *(uint4*)&ldsA[rr * 16 + c8 + 4] = a1;
        }
        __syncthreads();

        // load + dequant B (unscaled): two b128 loads exactly cover this lane's K set
        BF16x16 b;
        {
            const int4* bp = (const int4*)(Brow + (long)kb * 16);
            int4 w0 = bp[h];        // ints 4h..4h+3   -> K = 8h..8h+7
            int4 w1 = bp[2 + h];    // ints 8+4h..11+4h-> K = 16+8h..16+8h+7
            unsigned v40 = (unsigned)(w0.x & 255) | ((unsigned)(w0.y & 255) << 8) |
                           ((unsigned)(w0.z & 255) << 16) | ((unsigned)(w0.w & 255) << 24);
            unsigned v41 = (unsigned)(w1.x & 255) | ((unsigned)(w1.y & 255) << 8) |
                           ((unsigned)(w1.z & 255) << 16) | ((unsigned)(w1.w & 255) << 24);
            dq4(v40, &b.u[0]);
            dq4(v41, &b.u[4]);
        }
        float sc = ldexpf(1.0f, Srow[kb] - 127);

        #pragma unroll
        for (int mt = 0; mt < 8; ++mt) {
            BF16x16 a;
            const uint4* ap = (const uint4*)&ldsA[(mt * 16 + ln) * 16];
            a.q[0] = ap[h];         // K = 8h..8h+7
            a.q[1] = ap[2 + h];     // K = 16+8h..16+8h+7
            v8f part = __builtin_amdgcn_wmma_f32_16x16x32_bf16(
                false, a.v, false, b.v, (short)0, zf, false, false);
            #pragma unroll
            for (int i = 0; i < 8; ++i)
                acc[mt][i] = fmaf(sc, part[i], acc[mt][i]);
        }
    }

    if (j < N) {
        float bs = bias[(long)e * N + j];
        #pragma unroll
        for (int mt = 0; mt < 8; ++mt) {
            #pragma unroll
            for (int r8 = 0; r8 < 8; ++r8) {
                int s = mtile * 128 + mt * 16 + r8 + 8 * h;   // D: row = r8+8*half
                if (s < nE)
                    out[(long)(offe + s) * N + j] = acc[mt][r8] + bs;
            }
        }
    }
}

// ---------------- kernel 4: SwiGLU + MXFP8 qdq of activations ----------------
__global__ void swiglu_kernel(const float* __restrict__ hbuf, unsigned short* __restrict__ act) {
    int lane = threadIdx.x & 31;
    long blk = (long)blockIdx.x * (blockDim.x >> 5) + (threadIdx.x >> 5);
    if (blk >= (long)NA * (I_ / 32)) return;
    long row = blk / (I_ / 32);
    int  ib  = (int)(blk % (I_ / 32));
    long i   = (long)ib * 32 + lane;
    const float2* h2 = (const float2*)(hbuf + row * (long)N1);
    float2 gu = h2[i];
    float g = fminf(gu.x, 7.0f);
    float u = fminf(fmaxf(gu.y, -7.0f), 7.0f);
    float glu = g / (1.0f + expf(-1.702f * g));
    float a = (u + 1.0f) * glu;
    float m = fabsf(a);
    for (int d = 16; d > 0; d >>= 1) m = fmaxf(m, __shfl_xor(m, d, 32));
    float scale = mx_scale(m);
    float q = qdq_e4m3(a / scale) * scale;
    act[row * (long)I_ + i] = f32_to_bf16_bits(q);
}

// ---------------- kernel 6: weighted combine (deterministic per token) ----------------
__global__ void combine_kernel(const float* __restrict__ oa, const float* __restrict__ rw,
                               const int* __restrict__ slotOf, float* __restrict__ out) {
    long idx = (long)blockIdx.x * blockDim.x + threadIdx.x;
    if (idx >= (long)T_ * H_) return;
    long t = idx / H_;
    int  hc = (int)(idx % H_);
    int s0 = slotOf[t * 2], s1 = slotOf[t * 2 + 1];
    out[idx] = rw[t * 2]     * oa[(long)s0 * H_ + hc] +
               rw[t * 2 + 1] * oa[(long)s1 * H_ + hc];
}

// ---------------- launcher ----------------
extern "C" void kernel_launch(void* const* d_in, const int* in_sizes, int n_in,
                              void* d_out, int out_size, void* d_ws, size_t ws_size,
                              hipStream_t stream) {
    const float* x      = (const float*)d_in[0];
    const int*   eidx   = (const int*)  d_in[1];
    const float* rw     = (const float*)d_in[2];
    const int*   gub    = (const int*)  d_in[3];
    const int*   gus    = (const int*)  d_in[4];
    const float* gubias = (const float*)d_in[5];
    const int*   dnb    = (const int*)  d_in[6];
    const int*   dns    = (const int*)  d_in[7];
    const float* dnbias = (const float*)d_in[8];
    float* out = (float*)d_out;

    char* ws = (char*)d_ws;
    int* counts = (int*)(ws + OFF_COUNTS);
    int* offs   = (int*)(ws + OFF_OFFS);
    int* tok    = (int*)(ws + OFF_TOK);
    int* slot   = (int*)(ws + OFF_SLOT);
    unsigned short* xq  = (unsigned short*)(ws + OFF_XQ);
    float*          hb  = (float*)(ws + OFF_H);
    unsigned short* act = (unsigned short*)(ws + OFF_ACT);
    float*          oa  = (float*)(ws + OFF_OA);

    route_kernel<<<1, 1024, 0, stream>>>(eidx, counts, offs, tok, slot);

    qdqx_kernel<<<(T_ * NKB + 7) / 8, 256, 0, stream>>>(x, xq);

    // gate_up GEMM: [nE x 2880] @ [5760 x 2880]^T -> h
    moe_gemm<<<dim3(N1 / 128, (NA + 127) / 128, E_), 256, 0, stream>>>(
        xq, tok, offs, counts, gub, gus, gubias, hb, N1, H_, 1);

    swiglu_kernel<<<(NA * (I_ / 32) + 7) / 8, 256, 0, stream>>>(hb, act);

    // down GEMM: [nE x 2880] @ [2880 x 2880]^T -> oa
    moe_gemm<<<dim3((H_ + 127) / 128, (NA + 127) / 128, E_), 256, 0, stream>>>(
        act, tok, offs, counts, dnb, dns, dnbias, oa, H_, I_, 0);

    combine_kernel<<<(int)(((long)T_ * H_ + 255) / 256), 256, 0, stream>>>(oa, rw, slot, out);
    (void)in_sizes; (void)n_in; (void)out_size; (void)ws_size;
}
